// MambaStack_35828617183483
// MI455X (gfx1250) — matrix operand order repaired and checked by reading, not compiled, not run on hardware
//
#include <hip/hip_runtime.h>

// ---------------------------------------------------------------------------
// Mamba stack (L=4) for MI455X / gfx1250, wave32 + WMMA bf16.
// GEMMs (in_proj / x_proj / out_proj) run on v_wmma_f32_16x16x32_bf16 with
// f32 accumulation, double-buffered fragment loads, 8 WMMA per k-step/wave.
// conv/softplus/scan/layernorm are f32 VALU kernels.
// ---------------------------------------------------------------------------

#define LAYERS   4
#define D_MODEL  768
#define D_STATE  16
#define D_CONV   4
#define D_INNER  1536
#define DT_RANK  48
#define BB       2
#define TT       2048
#define BT       (BB * TT)          // 4096 rows
#define DBL_N    (DT_RANK + 2 * D_STATE)   // 80
#define LN_EPS   1e-5f

typedef __bf16 bf16;
typedef bf16  v16bf __attribute__((ext_vector_type(16)));
typedef float v8f   __attribute__((ext_vector_type(8)));

struct alignas(16) U4 { unsigned int x, y, z, w; };
union Frag { v16bf v; U4 q[2]; };

// ---------------------------------------------------------------------------
// f32 -> bf16 convert
// ---------------------------------------------------------------------------
__global__ __launch_bounds__(256) void f32_to_bf16_kernel(
    const float* __restrict__ in, bf16* __restrict__ out, int n)
{
    int i = blockIdx.x * 256 + threadIdx.x;
    if (i < n) out[i] = (bf16)in[i];
}

// ---------------------------------------------------------------------------
// WMMA GEMM:  C[M,N] (f32, ldc=N) = A[M,K] (bf16 row-major) * W[N,K]^T (bf16)
// Block: 256 threads = 8 waves arranged 4(M) x 2(N); block tile 128 x 128;
// wave tile 32 x 64 = 8 accumulators; K stepped by 64 with two fragment
// sets software-pipelined. M multiple of 128, K multiple of 64; N guarded by
// clamping B rows (garbage columns are never stored).
// ---------------------------------------------------------------------------
__global__ __launch_bounds__(256) void gemm_bf16_nt_kernel(
    const bf16* __restrict__ A, const bf16* __restrict__ W,
    float* __restrict__ C, int M, int N, int K)
{
    const int lane = threadIdx.x & 31;
    const int wave = threadIdx.x >> 5;
    const int wm   = wave & 3;          // 0..3  (M direction)
    const int wn   = wave >> 2;         // 0..1  (N direction)
    const int half = lane >> 4;         // 0/1
    const int l15  = lane & 15;

    const int m0 = blockIdx.y * 128 + wm * 32;
    const int n0 = blockIdx.x * 128 + wn * 64;

    v8f acc[2][4] = {};

    // A fragment row pointers (rows always in range: M multiple of 128)
    const bf16* pa[2];
    pa[0] = A + (size_t)(m0 + l15)      * K + half * 8;
    pa[1] = A + (size_t)(m0 + 16 + l15) * K + half * 8;

    // B(=W) fragment row pointers, clamped into range (no divergence)
    const bf16* pb[4];
    int cn[4];
    #pragma unroll
    for (int j = 0; j < 4; ++j) {
        int nb = n0 + j * 16 + l15;
        cn[j]  = nb;
        int nbc = (nb < N) ? nb : (N - 1);
        pb[j]  = W + (size_t)nbc * K + half * 8;
    }

    Frag a[2][2], b[2][4];      // [set][frag]

    // preload set 0 @ k = 0
    #pragma unroll
    for (int i = 0; i < 2; ++i) {
        a[0][i].q[0] = *(const U4*)(pa[i]);
        a[0][i].q[1] = *(const U4*)(pa[i] + 16);
    }
    #pragma unroll
    for (int j = 0; j < 4; ++j) {
        b[0][j].q[0] = *(const U4*)(pb[j]);
        b[0][j].q[1] = *(const U4*)(pb[j] + 16);
    }

    for (int k = 0; k < K; k += 64) {
        // load set 1 @ k+32 (always valid: K multiple of 64)
        #pragma unroll
        for (int i = 0; i < 2; ++i) {
            a[1][i].q[0] = *(const U4*)(pa[i] + k + 32);
            a[1][i].q[1] = *(const U4*)(pa[i] + k + 48);
        }
        #pragma unroll
        for (int j = 0; j < 4; ++j) {
            b[1][j].q[0] = *(const U4*)(pb[j] + k + 32);
            b[1][j].q[1] = *(const U4*)(pb[j] + k + 48);
        }
        if (k + 256 < K) {
            __builtin_prefetch(pa[0] + k + 256, 0, 3);
            __builtin_prefetch(pb[0] + k + 256, 0, 3);
        }

        // WMMA on set 0
        #pragma unroll
        for (int i = 0; i < 2; ++i)
            #pragma unroll
            for (int j = 0; j < 4; ++j)
                acc[i][j] = __builtin_amdgcn_wmma_f32_16x16x32_bf16(
                    false, a[0][i].v, false, b[0][j].v, (short)0,
                    acc[i][j], false, false);

        // load set 0 @ k+64
        if (k + 64 < K) {
            #pragma unroll
            for (int i = 0; i < 2; ++i) {
                a[0][i].q[0] = *(const U4*)(pa[i] + k + 64);
                a[0][i].q[1] = *(const U4*)(pa[i] + k + 80);
            }
            #pragma unroll
            for (int j = 0; j < 4; ++j) {
                b[0][j].q[0] = *(const U4*)(pb[j] + k + 64);
                b[0][j].q[1] = *(const U4*)(pb[j] + k + 80);
            }
        }

        // WMMA on set 1
        #pragma unroll
        for (int i = 0; i < 2; ++i)
            #pragma unroll
            for (int j = 0; j < 4; ++j)
                acc[i][j] = __builtin_amdgcn_wmma_f32_16x16x32_bf16(
                    false, a[1][i].v, false, b[1][j].v, (short)0,
                    acc[i][j], false, false);
    }

    // D layout: lane col = l15; rows = half*8 + r (and +16 for second A frag)
    #pragma unroll
    for (int j = 0; j < 4; ++j) {
        if (cn[j] < N) {
            #pragma unroll
            for (int r = 0; r < 8; ++r) {
                int m = m0 + half * 8 + r;
                C[(size_t)m * N + cn[j]]        = acc[0][j][r];
                C[(size_t)(m + 16) * N + cn[j]] = acc[1][j][r];
            }
        }
    }
}

// ---------------------------------------------------------------------------
// Causal depthwise conv1d (k=4) + SiLU.  xin = xz[..., 0:D_INNER].
// Writes xc (f32) and xc_bf (bf16) for the next GEMM.
// ---------------------------------------------------------------------------
__global__ __launch_bounds__(256) void conv_silu_kernel(
    const float* __restrict__ xz, const float* __restrict__ convw,
    const float* __restrict__ convb, float* __restrict__ xc,
    bf16* __restrict__ xcbf)
{
    int idx = blockIdx.x * 256 + threadIdx.x;     // over BT*D_INNER
    if (idx >= BT * D_INNER) return;
    int d   = idx % D_INNER;
    int row = idx / D_INNER;                      // b*T + t
    int t   = row % TT;

    float acc = convb[d];
    #pragma unroll
    for (int k = 0; k < D_CONV; ++k) {
        int dtk = k - (D_CONV - 1);
        if (t + dtk >= 0)
            acc += xz[(size_t)(row + dtk) * (2 * D_INNER) + d] * convw[d * D_CONV + k];
    }
    float s = acc / (1.f + __expf(-acc));         // SiLU
    xc[idx]   = s;
    xcbf[idx] = (bf16)s;
}

// ---------------------------------------------------------------------------
// dt = softplus(dbl[:, :48] @ dtw^T + dtb)   (K=48 -> plain VALU dot)
// ---------------------------------------------------------------------------
__global__ __launch_bounds__(256) void dt_softplus_kernel(
    const float* __restrict__ dbl, const float* __restrict__ dtw,
    const float* __restrict__ dtb, float* __restrict__ dt)
{
    int idx = blockIdx.x * 256 + threadIdx.x;     // over BT*D_INNER
    if (idx >= BT * D_INNER) return;
    int d   = idx % D_INNER;
    int row = idx / D_INNER;

    const float* in = dbl + (size_t)row * DBL_N;
    const float* w  = dtw + (size_t)d * DT_RANK;
    float acc = dtb[d];
    #pragma unroll 8
    for (int r = 0; r < DT_RANK; ++r) acc += in[r] * w[r];
    dt[idx] = (acc > 20.f) ? acc : log1pf(__expf(acc));
}

// ---------------------------------------------------------------------------
// Selective scan: one lane per (b, d, n); 16 lanes/channel, shfl_xor reduce
// over the 16 states each step. Fuses D-skip + SiLU(z) gate; writes bf16 y.
// ---------------------------------------------------------------------------
__global__ __launch_bounds__(256) void scan_kernel(
    const float* __restrict__ dt, const float* __restrict__ xc,
    const float* __restrict__ xz, const float* __restrict__ dbl,
    const float* __restrict__ A_log, const float* __restrict__ Dp,
    bf16* __restrict__ ybf)
{
    int gid   = blockIdx.x * 256 + threadIdx.x;
    int n     = gid & 15;
    int group = gid >> 4;                          // (b, d) channel
    int b     = group / D_INNER;
    int d     = group - b * D_INNER;

    float Adn = -__expf(A_log[(size_t)d * D_STATE + n]);
    float Dd  = Dp[d];
    float h   = 0.f;

    for (int t = 0; t < TT; ++t) {
        size_t row = (size_t)b * TT + t;
        float dtv = dt[row * D_INNER + d];
        float xcv = xc[row * D_INNER + d];
        float Bv  = dbl[row * DBL_N + DT_RANK + n];
        float Cv  = dbl[row * DBL_N + DT_RANK + D_STATE + n];
        h = __expf(dtv * Adn) * h + (dtv * xcv) * Bv;
        float p = h * Cv;
        p += __shfl_xor(p, 1);
        p += __shfl_xor(p, 2);
        p += __shfl_xor(p, 4);
        p += __shfl_xor(p, 8);
        if (n == 0) {
            float zv = xz[row * (2 * D_INNER) + D_INNER + d];
            float y  = (p + xcv * Dd) * (zv / (1.f + __expf(-zv)));
            ybf[row * D_INNER + d] = (bf16)y;
        }
    }
}

// ---------------------------------------------------------------------------
// LayerNorm(h) * g + b + residual  (one block per row of 768)
// ---------------------------------------------------------------------------
__global__ __launch_bounds__(256) void ln_residual_kernel(
    const float* __restrict__ h, const float* __restrict__ res,
    const float* __restrict__ g, const float* __restrict__ bb,
    float* __restrict__ out)
{
    __shared__ float red[256];
    int row = blockIdx.x;
    int tid = threadIdx.x;
    const float* hr = h + (size_t)row * D_MODEL;

    float v0 = hr[tid], v1 = hr[tid + 256], v2 = hr[tid + 512];
    red[tid] = v0 + v1 + v2;
    __syncthreads();
    for (int s = 128; s > 0; s >>= 1) {
        if (tid < s) red[tid] += red[tid + s];
        __syncthreads();
    }
    float mu = red[0] / (float)D_MODEL;
    __syncthreads();

    float d0 = v0 - mu, d1 = v1 - mu, d2 = v2 - mu;
    red[tid] = d0 * d0 + d1 * d1 + d2 * d2;
    __syncthreads();
    for (int s = 128; s > 0; s >>= 1) {
        if (tid < s) red[tid] += red[tid + s];
        __syncthreads();
    }
    float rs = rsqrtf(red[0] / (float)D_MODEL + LN_EPS);

    const float* rr = res + (size_t)row * D_MODEL;
    float* o = out + (size_t)row * D_MODEL;
    o[tid]       = d0 * rs * g[tid]       + bb[tid]       + rr[tid];
    o[tid + 256] = d1 * rs * g[tid + 256] + bb[tid + 256] + rr[tid + 256];
    o[tid + 512] = d2 * rs * g[tid + 512] + bb[tid + 512] + rr[tid + 512];
}

// ---------------------------------------------------------------------------
// Host driver
// ---------------------------------------------------------------------------
static inline int cdiv(int a, int b) { return (a + b - 1) / b; }

extern "C" void kernel_launch(void* const* d_in, const int* in_sizes, int n_in,
                              void* d_out, int out_size, void* d_ws, size_t ws_size,
                              hipStream_t stream) {
    (void)in_sizes; (void)n_in; (void)out_size; (void)ws_size;

    const float* x_in    = (const float*)d_in[0];
    const float* inw     = (const float*)d_in[1];   // [L, 3072, 768]
    const float* convw   = (const float*)d_in[2];   // [L, 1536, 4]
    const float* convb   = (const float*)d_in[3];   // [L, 1536]
    const float* xpw     = (const float*)d_in[4];   // [L, 80, 1536]
    const float* dtw     = (const float*)d_in[5];   // [L, 1536, 48]
    const float* dtb     = (const float*)d_in[6];   // [L, 1536]
    const float* A_log   = (const float*)d_in[7];   // [L, 1536, 16]
    const float* Dp      = (const float*)d_in[8];   // [L, 1536]
    const float* outw    = (const float*)d_in[9];   // [L, 768, 1536]
    const float* ln_g    = (const float*)d_in[10];  // [L, 768]
    const float* ln_b    = (const float*)d_in[11];  // [L, 768]

    // ---- workspace carve-up ----
    char* ws = (char*)d_ws;
    size_t off = 0;
    auto carve = [&](size_t bytes) { size_t o = off; off = (off + bytes + 255) & ~(size_t)255; return o; };

    float* xA   = (float*)(ws + carve((size_t)BT * D_MODEL * 4));
    float* xB   = (float*)(ws + carve((size_t)BT * D_MODEL * 4));
    bf16*  xbf  = (bf16*) (ws + carve((size_t)BT * D_MODEL * 2));
    bf16*  wbf  = (bf16*) (ws + carve((size_t)(2 * D_INNER) * D_MODEL * 2)); // biggest weight
    float* xz   = (float*)(ws + carve((size_t)BT * 2 * D_INNER * 4));
    float* xc   = (float*)(ws + carve((size_t)BT * D_INNER * 4));            // also reused as h
    bf16*  xcbf = (bf16*) (ws + carve((size_t)BT * D_INNER * 2));            // also y (bf16)
    float* dbl  = (float*)(ws + carve((size_t)BT * DBL_N * 4));
    float* dtbuf= (float*)(ws + carve((size_t)BT * D_INNER * 4));

    const float* cur = x_in;
    for (int l = 0; l < LAYERS; ++l) {
        float* nxt = (l == LAYERS - 1) ? (float*)d_out : ((l & 1) ? xA : xB);

        // 1) convert layer input + in_proj weight to bf16
        f32_to_bf16_kernel<<<cdiv(BT * D_MODEL, 256), 256, 0, stream>>>(
            cur, xbf, BT * D_MODEL);
        f32_to_bf16_kernel<<<cdiv(2 * D_INNER * D_MODEL, 256), 256, 0, stream>>>(
            inw + (size_t)l * 2 * D_INNER * D_MODEL, wbf, 2 * D_INNER * D_MODEL);

        // 2) xz = x @ in_proj^T : [4096, 3072]
        gemm_bf16_nt_kernel<<<dim3(cdiv(2 * D_INNER, 128), BT / 128), 256, 0, stream>>>(
            xbf, wbf, xz, BT, 2 * D_INNER, D_MODEL);

        // 3) depthwise causal conv + SiLU -> xc, xcbf
        conv_silu_kernel<<<cdiv(BT * D_INNER, 256), 256, 0, stream>>>(
            xz, convw + (size_t)l * D_INNER * D_CONV, convb + (size_t)l * D_INNER,
            xc, xcbf);

        // 4) dbl = xc @ x_proj^T : [4096, 80]
        f32_to_bf16_kernel<<<cdiv(DBL_N * D_INNER, 256), 256, 0, stream>>>(
            xpw + (size_t)l * DBL_N * D_INNER, wbf, DBL_N * D_INNER);
        gemm_bf16_nt_kernel<<<dim3(cdiv(DBL_N, 128), BT / 128), 256, 0, stream>>>(
            xcbf, wbf, dbl, BT, DBL_N, D_INNER);

        // 5) dt = softplus(dbl[:, :48] @ dt_proj^T + dtb)
        dt_softplus_kernel<<<cdiv(BT * D_INNER, 256), 256, 0, stream>>>(
            dbl, dtw + (size_t)l * D_INNER * DT_RANK, dtb + (size_t)l * D_INNER, dtbuf);

        // 6) selective scan + D-skip + SiLU(z) gate -> y (bf16 into xcbf)
        scan_kernel<<<(BB * D_INNER * D_STATE) / 256, 256, 0, stream>>>(
            dtbuf, xc, xz, dbl,
            A_log + (size_t)l * D_INNER * D_STATE, Dp + (size_t)l * D_INNER, xcbf);

        // 7) h = y @ out_proj^T : [4096, 768]  (h aliases xc, now dead)
        f32_to_bf16_kernel<<<cdiv(D_MODEL * D_INNER, 256), 256, 0, stream>>>(
            outw + (size_t)l * D_MODEL * D_INNER, wbf, D_MODEL * D_INNER);
        gemm_bf16_nt_kernel<<<dim3(cdiv(D_MODEL, 128), BT / 128), 256, 0, stream>>>(
            xcbf, wbf, xc, BT, D_MODEL, D_INNER);

        // 8) x = LayerNorm(h)*g + b + residual
        ln_residual_kernel<<<BT, 256, 0, stream>>>(
            xc, cur, ln_g + (size_t)l * D_MODEL, ln_b + (size_t)l * D_MODEL, nxt);

        cur = nxt;
    }
}
